// LSTM_36910948942371
// MI455X (gfx1250) — matrix operand order repaired and checked by reading, not compiled, not run on hardware
//
#include <hip/hip_runtime.h>

typedef __attribute__((ext_vector_type(16))) _Float16 v16h;
typedef __attribute__((ext_vector_type(8)))  _Float16 v8h;
typedef __attribute__((ext_vector_type(8)))  float    v8f;

#if defined(__has_builtin)
#if __has_builtin(__builtin_amdgcn_tensor_load_to_lds) && \
    __has_builtin(__builtin_amdgcn_s_wait_tensorcnt)
#define HAVE_TDM 1
#endif
#endif

#ifdef HAVE_TDM
typedef __attribute__((ext_vector_type(4))) unsigned int u32x4;
typedef __attribute__((ext_vector_type(8))) int          i32x8;
typedef __attribute__((ext_vector_type(4))) int          i32x4;

// Issue one TDM 2D tile load: rows x cols f16 tile at gsrc (row pitch = pitch
// elements) -> LDS byte offset lds_off.  D# packing per CDNA5 ISA ch.8.
// 6-arg builtin form (clang-23 / therock-10.0 headers).
static __device__ __forceinline__ void tdm_load_tile(const _Float16* gsrc,
                                                     unsigned lds_off,
                                                     int rows, int cols,
                                                     int pitch, int theight) {
    unsigned long long ga = (unsigned long long)(size_t)gsrc;
    u32x4 g0;
    g0[0] = 1u;                                        // count=1, user descriptor
    g0[1] = lds_off;                                   // lds_addr
    g0[2] = (unsigned)(ga & 0xFFFFFFFFu);              // global_addr[31:0]
    g0[3] = (unsigned)((ga >> 32) & 0x1FFFFFFu) | (2u << 30);  // [56:32] | type=2
    i32x8 g1;
    g1[0] = (1 << 16);                                 // data_size=1 (2B), mask=0
    g1[1] = (int)(((unsigned)pitch & 0xFFFFu) << 16);  // tensor_dim0[15:0]
    g1[2] = (int)((((unsigned)pitch >> 16) & 0xFFFFu) |
                  (((unsigned)theight & 0xFFFFu) << 16));
    g1[3] = (int)((((unsigned)theight >> 16) & 0xFFFFu) |
                  (((unsigned)cols & 0xFFFFu) << 16)); // tile_dim0 = cols
    g1[4] = rows & 0xFFFF;                             // tile_dim1, tile_dim2=0
    g1[5] = pitch;                                     // tensor_dim0_stride[31:0]
    g1[6] = 0;
    g1[7] = 0;
    i32x4 z4 = {};
    i32x8 z8 = {};
    __builtin_amdgcn_tensor_load_to_lds(g0, g1, z4, z4, z8, 0);
}
#endif

// ---------------------------------------------------------------------------
// WMMA helpers (CDNA5 wave32, v_wmma_f32_16x16x32_f16)
// ---------------------------------------------------------------------------
static __device__ __forceinline__ v8f wmma_f16(v16h a, v16h b, v8f c) {
    return __builtin_amdgcn_wmma_f32_16x16x32_f16(false, a, false, b, (short)0, c,
                                                  false, false);
}

// A fragment: 16x32 f16 tile, row-major source.
static __device__ __forceinline__ v16h load_a_frag(const _Float16* A, int lda,
                                                   int m0, int k0, int lane) {
    int row = m0 + (lane & 15);
    int hl  = lane >> 4;
    const _Float16* p = A + (size_t)row * lda + k0 + 16 * hl;
    union { v16h v; v8h h[2]; } a;
    a.h[0] = *(const v8h*)(p);
    a.h[1] = *(const v8h*)(p + 8);
    return a.v;
}

// B fragment: 32x16 tile from pre-transposed Bt[N][K] row-major.
static __device__ __forceinline__ v16h load_b_frag(const _Float16* Bt, int ldb,
                                                   int n0, int k0, int lane) {
    int col = n0 + (lane & 15);
    int hl  = lane >> 4;
    return *(const v16h*)(Bt + (size_t)col * ldb + k0 + 16 * hl);
}

static __device__ __forceinline__ float sigmoid_f(float x) {
    return 1.0f / (1.0f + __expf(-x));
}

// ---------------------------------------------------------------------------
// Prep kernels
// ---------------------------------------------------------------------------
__global__ void prep_xemb(const int* __restrict__ X, const float* __restrict__ emb,
                          _Float16* __restrict__ Xemb, int total) {
    int idx = blockIdx.x * blockDim.x + threadIdx.x;
    if (idx >= total) return;
    int bt = idx / 192;
    int e  = idx - bt * 192;
    Xemb[idx] = (_Float16)emb[(size_t)X[bt] * 192 + e];
}

__global__ void prep_transpose(const float* __restrict__ W, _Float16* __restrict__ Wt,
                               int rows, int cols) {
    int idx = blockIdx.x * blockDim.x + threadIdx.x;
    int total = rows * cols;
    if (idx >= total) return;
    int n = idx / rows;
    int k = idx - n * rows;
    Wt[idx] = (_Float16)W[(size_t)k * cols + n];
}

__global__ void prep_gates(const float* __restrict__ Wf, const float* __restrict__ Wi,
                           const float* __restrict__ Wg, const float* __restrict__ Wo,
                           const float* __restrict__ bf, const float* __restrict__ bi,
                           const float* __restrict__ bg, const float* __restrict__ bo,
                           _Float16* __restrict__ Wh_t, _Float16* __restrict__ Wx_t,
                           float* __restrict__ bias_g) {
    int idx = blockIdx.x * blockDim.x + threadIdx.x;
    if (idx >= 4 * 512 * 512) return;
    int gate = idx >> 18;
    int rem  = idx & 0x3FFFF;
    int j    = rem >> 9;
    int k    = rem & 511;
    const float* W = (gate == 0) ? Wf : (gate == 1) ? Wi : (gate == 2) ? Wg : Wo;
    int o = ((gate << 9) + j) * 512 + k;
    Wh_t[o] = (_Float16)W[(size_t)k * 512 + j];
    Wx_t[o] = (_Float16)W[(size_t)(512 + k) * 512 + j];
    if (k == 0) {
        const float* b = (gate == 0) ? bf : (gate == 1) ? bi : (gate == 2) ? bg : bo;
        bias_g[(gate << 9) + j] = b[j];
    }
}

__global__ void scan_init(_Float16* __restrict__ hglob, unsigned* __restrict__ ctr) {
    int i = blockIdx.x * blockDim.x + threadIdx.x;
    if (i == 0) *ctr = 0u;
    if (i < 32 * 512) hglob[i] = (_Float16)0.0f;
}

// ---------------------------------------------------------------------------
// WMMA GEMM: C[M,N] = A[M,K](f16) @ Bt[N,K]^T(f16) + bias.
// A block tile (64x32 per K-step) staged into LDS with the Tensor Data Mover
// (double buffered, s_wait_tensorcnt), B direct from global with prefetch.
//  EPI=0: f16 out; EPI=1: f32 out remapped (b,t)->(t*32+b); EPI=2: f32*scale.
// ---------------------------------------------------------------------------
template <int EPI>
__global__ void __launch_bounds__(256)
gemm_wmma(const _Float16* __restrict__ A, const _Float16* __restrict__ Bt,
          int M, int N, int K, int lda, int ldb,
          const float* __restrict__ bias, void* __restrict__ Cout, float scale) {
    int tid  = threadIdx.x;
    int lane = tid & 31;
    int wave = tid >> 5;
    int wm   = wave >> 2;
    int wn   = wave & 3;
    int m0   = blockIdx.x * 64 + wm * 32;
    int n0   = blockIdx.y * 128 + wn * 32;
    int hl   = lane >> 4;
    int lc   = lane & 15;

    v8f acc[2][2] = {};

#ifdef HAVE_TDM
    __shared__ _Float16 a_st[2][64 * 32];   // double-buffered A tile (8 KB)
    const _Float16* Ablk = A + (size_t)(blockIdx.x * 64) * lda;
    int nsteps = K / 32;
    if (wave == 0) tdm_load_tile(Ablk, 0u, 64, 32, lda, M);
    for (int s = 0; s < nsteps; ++s) {
        int k0  = s << 5;
        int cur = s & 1;
        __syncthreads();                     // buffer cur^1 free for reuse
        if (wave == 0) {
            if (s + 1 < nsteps) {
                tdm_load_tile(Ablk + (k0 + 32), (unsigned)((cur ^ 1) * 4096),
                              64, 32, lda, M);
                __builtin_amdgcn_s_wait_tensorcnt((short)1);  // tile `cur` done
            } else {
                __builtin_amdgcn_s_wait_tensorcnt((short)0);
            }
        }
        __syncthreads();                     // tile cur visible to all waves
        v16h a0 = load_a_frag(a_st[cur], 32, wm * 32,      0, lane);
        v16h a1 = load_a_frag(a_st[cur], 32, wm * 32 + 16, 0, lane);
        v16h b0 = load_b_frag(Bt, ldb, n0,      k0, lane);
        v16h b1 = load_b_frag(Bt, ldb, n0 + 16, k0, lane);
        if (s + 1 < nsteps) {
            __builtin_prefetch(Bt + (size_t)(n0 + lc) * ldb + k0 + 32, 0, 0);
            __builtin_prefetch(Bt + (size_t)(n0 + 16 + lc) * ldb + k0 + 32, 0, 0);
        }
        acc[0][0] = wmma_f16(a0, b0, acc[0][0]);
        acc[0][1] = wmma_f16(a0, b1, acc[0][1]);
        acc[1][0] = wmma_f16(a1, b0, acc[1][0]);
        acc[1][1] = wmma_f16(a1, b1, acc[1][1]);
    }
#else
    for (int k0 = 0; k0 < K; k0 += 32) {
        v16h a0 = load_a_frag(A, lda, m0,      k0, lane);
        v16h a1 = load_a_frag(A, lda, m0 + 16, k0, lane);
        v16h b0 = load_b_frag(Bt, ldb, n0,      k0, lane);
        v16h b1 = load_b_frag(Bt, ldb, n0 + 16, k0, lane);
        acc[0][0] = wmma_f16(a0, b0, acc[0][0]);
        acc[0][1] = wmma_f16(a0, b1, acc[0][1]);
        acc[1][0] = wmma_f16(a1, b0, acc[1][0]);
        acc[1][1] = wmma_f16(a1, b1, acc[1][1]);
    }
#endif

    for (int ti = 0; ti < 2; ++ti) {
        for (int tj = 0; tj < 2; ++tj) {
            for (int v = 0; v < 8; ++v) {
                float val = acc[ti][tj][v];
                int row = m0 + ti * 16 + v + 8 * hl;
                int col = n0 + tj * 16 + lc;
                if (EPI == 0) {
                    ((_Float16*)Cout)[(size_t)row * N + col] =
                        (_Float16)(val + bias[col]);
                } else if (EPI == 1) {
                    int b = row >> 9;
                    int t = row & 511;
                    ((float*)Cout)[(((size_t)t << 5) + b) * N + col] = val + bias[col];
                } else {
                    ((float*)Cout)[(size_t)row * N + col] = (val + bias[col]) * scale;
                }
            }
        }
    }
}

// ---------------------------------------------------------------------------
// LSTM scan across 16 workgroups (one j-slab of 32 columns x 4 gates each).
// Each WG parks its 128 KB slice of Wh in LDS for the whole scan; per step the
// only global traffic is the 32 KB h broadcast + 16 KB GX slice.  Cross-WG
// sync: cluster split-barrier (s_barrier_signal/-wait -3) when launched as a
// cluster, else a monotonic device-scope atomic counter.
// Wave w (of 16): mi=w&1, jt=(w>>1)&1, gate=w>>2 -> one 16x16 gate tile.
// Waves 0..3 fuse f/i/g/o elementwise (c state lives in their registers).
// ---------------------------------------------------------------------------
__global__ void __launch_bounds__(512)
lstm_scan(const _Float16* __restrict__ Wh_t, const float* __restrict__ GX,
          _Float16* __restrict__ hs, _Float16* __restrict__ hglob,
          unsigned* __restrict__ ctr) {
    __shared__ _Float16 wh_lds[128 * 512];   // 128 KB: this WG's Wh slice
    __shared__ _Float16 h_lds[32 * 512];     // 32 KB: h mirror (aliased by gbuf)
    float* gbuf = (float*)h_lds;             // 16 KB gate-tile exchange (aliased)

    int tid  = threadIdx.x;
    int lane = tid & 31;
    int wave = tid >> 5;
    int hl   = lane >> 4;
    int lc   = lane & 15;
    int wg   = blockIdx.x;                   // 0..15
    int jW   = wg << 5;                      // j-slab base

    int mi   = wave & 1;
    int jt   = (wave >> 1) & 1;
    int gate = wave >> 2;
    int nloc = gate * 32 + jt * 16;          // local B row base in wh_lds

#if defined(__has_builtin) && __has_builtin(__builtin_amdgcn_cluster_id_x)
    int incluster = __builtin_amdgcn_cluster_id_x();
#else
    int incluster = 0;
#endif

    // Park Wh slice in LDS: local row r = gate*32+jj  <->  global n = gate*512+jW+jj
    for (int i = tid * 8; i < 128 * 512; i += 512 * 8) {
        int r = i >> 9;
        int k = i & 511;
        int n = ((r >> 5) << 9) + jW + (r & 31);
        *(v8h*)&wh_lds[i] = *(const v8h*)(Wh_t + ((size_t)n << 9) + k);
    }
    for (int i = tid * 8; i < 32 * 512; i += 512 * 8)
        *(v8h*)&h_lds[i] = (v8h)(_Float16)0.0f;
    v8f c_reg = {};
    __syncthreads();

    #pragma unroll 1
    for (int t = 0; t < 512; ++t) {
        // Prefetch next step's GX slice (128 cache lines) while we compute.
        if (t + 1 < 512 && tid < 128) {
            int r = tid >> 2, g = tid & 3;
            __builtin_prefetch(GX + ((size_t)(t + 1) * 32 + r) * 2048 +
                               (g << 9) + jW, 0, 0);
        }

        // One 16x16 gate tile per wave:  h[32,512] @ Wh_slice
        v8f acc = {};
        #pragma unroll 4
        for (int k0 = 0; k0 < 512; k0 += 32) {
            v16h a = load_a_frag(h_lds, 512, mi * 16, k0, lane);   // ds_load
            v16h b = load_b_frag(wh_lds, 512, nloc, k0, lane);     // ds_load
            acc = wmma_f16(a, b, acc);
        }
        __syncthreads();                         // h reads done -> gbuf may alias

        #pragma unroll
        for (int v = 0; v < 8; ++v)
            gbuf[wave * 256 + (v + 8 * hl) * 16 + lc] = acc[v];
        __syncthreads();

        if (wave < 4) {                          // fuse gates, update c/h
            int mi2 = wave & 1;
            int jt2 = (wave >> 1) & 1;
            int base = mi2 + 2 * jt2;            // tile idx = base + 4*gate
            int jloc = jt2 * 16 + lc;
            #pragma unroll
            for (int v = 0; v < 8; ++v) {
                int e   = (v + 8 * hl) * 16 + lc;
                int row = mi2 * 16 + v + 8 * hl; // batch index
                const float* g = GX + ((size_t)t * 32 + row) * 2048 + jW + jloc;
                float f  = sigmoid_f(gbuf[(base + 0) * 256 + e] + g[0]);
                float ii = sigmoid_f(gbuf[(base + 4) * 256 + e] + g[512]);
                float gg = tanhf(gbuf[(base + 8) * 256 + e] + g[1024]);
                float o  = sigmoid_f(gbuf[(base + 12) * 256 + e] + g[1536]);
                float c  = c_reg[v] * f + ii * gg;
                c_reg[v] = c;
                float h  = tanhf(c) * o;
                int jcol = jW + jloc;
                hglob[row * 512 + jcol] = (_Float16)h;
                hs[((size_t)row * 512 + t) * 512 + jcol] = (_Float16)h;
            }
        }
        __threadfence();                         // h slice visible device-wide
        __syncthreads();

        // ---- cross-workgroup step barrier ----
        if (incluster != 0) {
            if (wave == 0) asm volatile("s_barrier_signal -3" ::: "memory");
            asm volatile("s_barrier_wait -3" ::: "memory");
        } else if (tid == 0) {
            __hip_atomic_fetch_add(ctr, 1u, __ATOMIC_RELEASE,
                                   __HIP_MEMORY_SCOPE_AGENT);
            unsigned need = 16u * (unsigned)(t + 1);
            while (__hip_atomic_load(ctr, __ATOMIC_ACQUIRE,
                                     __HIP_MEMORY_SCOPE_AGENT) < need)
                __builtin_amdgcn_s_sleep(1);
        }
        __syncthreads();

        // Refresh local h mirror (overwrites gbuf alias - safe now).
        for (int i = tid * 8; i < 32 * 512; i += 512 * 8)
            *(v8h*)&h_lds[i] = *(const v8h*)(hglob + i);
        __syncthreads();
    }
}

// ---------------------------------------------------------------------------
// Workspace layout (bytes, 256-aligned)
// ---------------------------------------------------------------------------
#define OFF_XEMB   ((size_t)0)
#define OFF_WINT   ((size_t)6291456)
#define OFF_WX     ((size_t)6488064)
#define OFF_WH     ((size_t)8585216)
#define OFF_BIASG  ((size_t)10682368)
#define OFF_WHEAD  ((size_t)10690560)
#define OFF_XH     ((size_t)19079168)
#define OFF_HS     ((size_t)35856384)
#define OFF_GX     ((size_t)52633600)
#define OFF_HG     ((size_t)186851328)
#define OFF_CTR    ((size_t)186884096)

extern "C" void kernel_launch(void* const* d_in, const int* in_sizes, int n_in,
                              void* d_out, int out_size, void* d_ws, size_t ws_size,
                              hipStream_t stream) {
    const int*   X      = (const int*)  d_in[0];
    const float* emb    = (const float*)d_in[1];
    const float* W_in   = (const float*)d_in[2];
    const float* b_in   = (const float*)d_in[3];
    const float* Wf     = (const float*)d_in[4];
    const float* bf     = (const float*)d_in[5];
    const float* Wi     = (const float*)d_in[6];
    const float* bi     = (const float*)d_in[7];
    const float* Wg     = (const float*)d_in[8];
    const float* bg     = (const float*)d_in[9];
    const float* Wo     = (const float*)d_in[10];
    const float* bo     = (const float*)d_in[11];
    const float* W_head = (const float*)d_in[12];
    const float* b_head = (const float*)d_in[13];

    char* ws = (char*)d_ws;
    _Float16* Xemb     = (_Float16*)(ws + OFF_XEMB);
    _Float16* W_in_t   = (_Float16*)(ws + OFF_WINT);
    _Float16* Wx_t     = (_Float16*)(ws + OFF_WX);
    _Float16* Wh_t     = (_Float16*)(ws + OFF_WH);
    float*    bias_g   = (float*)   (ws + OFF_BIASG);
    _Float16* W_head_t = (_Float16*)(ws + OFF_WHEAD);
    _Float16* x_h      = (_Float16*)(ws + OFF_XH);
    _Float16* hs_h     = (_Float16*)(ws + OFF_HS);
    float*    GX       = (float*)   (ws + OFF_GX);
    _Float16* hglob    = (_Float16*)(ws + OFF_HG);
    unsigned* ctr      = (unsigned*)(ws + OFF_CTR);

    prep_xemb<<<(16384 * 192 + 255) / 256, 256, 0, stream>>>(X, emb, Xemb, 16384 * 192);
    prep_transpose<<<(192 * 512 + 255) / 256, 256, 0, stream>>>(W_in, W_in_t, 192, 512);
    prep_transpose<<<(512 * 8192 + 255) / 256, 256, 0, stream>>>(W_head, W_head_t, 512, 8192);
    prep_gates<<<(4 * 512 * 512 + 255) / 256, 256, 0, stream>>>(
        Wf, Wi, Wg, Wo, bf, bi, bg, bo, Wh_t, Wx_t, bias_g);

    // x = emb[X] @ W_in + b_in   [16384,512] f16
    gemm_wmma<0><<<dim3(16384 / 64, 512 / 128), 256, 0, stream>>>(
        Xemb, W_in_t, 16384, 512, 192, 192, 192, b_in, x_h, 1.0f);

    // GX = x @ [Wfx|Wix|Wgx|Wox] + biases, stored [T,B,2048] f32
    gemm_wmma<1><<<dim3(16384 / 64, 2048 / 128), 256, 0, stream>>>(
        x_h, Wx_t, 16384, 2048, 512, 512, 512, bias_g, GX, 1.0f);

    // recurrent scan across 16 cooperating workgroups
    scan_init<<<64, 256, 0, stream>>>(hglob, ctr);
    lstm_scan<<<16, 512, 0, stream>>>(Wh_t, GX, hs_h, hglob, ctr);

    // logits = (hs @ W_head + b_head) * 0.1   [16384,8192] f32
    gemm_wmma<2><<<dim3(16384 / 64, 8192 / 128), 256, 0, stream>>>(
        hs_h, W_head_t, 16384, 8192, 512, 512, 512, b_head, (float*)d_out, 0.1f);
}
